// MaskClusterAttention_1889785610594
// MI455X (gfx1250) — matrix operand order repaired and checked by reading, not compiled, not run on hardware
//
#include <hip/hip_runtime.h>

// MaskClusterAttention for gfx1250 (MI455X): f16 WMMA everywhere, fp32 accumulate.
// N=4096 tokens, D=1024, H=16 heads, hd=64.

typedef __attribute__((ext_vector_type(16))) _Float16 v16h;
typedef __attribute__((ext_vector_type(8)))  _Float16 v8h;
typedef __attribute__((ext_vector_type(8)))  float    v8f;

#define N_TOK 4096
#define DIM   1024
#define NH    16
#define HD    64
#define TD    3072   // 3*DIM

static __device__ __forceinline__ v8f wmma_f16(v16h a, v16h b, v8f c) {
  // D = A(16x32) * B(32x16) + C, f32 accumulate
  return __builtin_amdgcn_wmma_f32_16x16x32_f16(false, a, false, b, (short)0, c,
                                                false, false);
}

static __device__ __forceinline__ v16h pack16(v8h lo, v8h hi) {
  v16h r;
#pragma unroll
  for (int i = 0; i < 8; ++i) { r[i] = lo[i]; r[i + 8] = hi[i]; }
  return r;
}

// ---------------------------------------------------------------- f32 -> f16
__global__ void k_cvt(const float* __restrict__ s, _Float16* __restrict__ d, int n) {
  int i = blockIdx.x * blockDim.x + threadIdx.x;
  int stride = gridDim.x * blockDim.x;
  for (; i < n; i += stride) d[i] = (_Float16)s[i];
}

// ------------------------------------------------------- WMMA GEMM: A * W^T + b
// A: [M, kdim] f16 row-major. W: [Nout, kdim] f16 row-major.
// Wave tile: 32(M) x 64(N): 2 A-fragments x 4 B-fragments, 8 wmma / 32-K step.
// A-fragment: lane m = L&15, half h = L>>4, elems j = K h*16+j -> 16 contiguous halves
//             of A row m (one 32B load).
// B-fragment (= W^T): lane n = L&15, elems j = K h*16+j -> 16 contiguous halves
//             of W row n (one 32B load).
__global__ __launch_bounds__(256)
void k_gemm(const _Float16* __restrict__ A, int lda,
            const _Float16* __restrict__ W, int ldw,
            const float* __restrict__ bias,
            _Float16* __restrict__ outh, float* __restrict__ outf, int ldo,
            int kdim) {
  const int lane = threadIdx.x & 31;
  const int wave = threadIdx.x >> 5;
  const int m0 = blockIdx.x * 32;
  const int n0 = blockIdx.y * 512 + wave * 64;
  const int ml = lane & 15;
  const int h  = lane >> 4;

  v8f acc[2][4] = {};

  for (int k0 = 0; k0 < kdim; k0 += 32) {
    v16h av[2];
#pragma unroll
    for (int u = 0; u < 2; ++u) {
      const _Float16* ap = A + (size_t)(m0 + u * 16 + ml) * lda + k0;
      av[u] = *(const v16h*)(ap + h * 16);
    }
#pragma unroll
    for (int t = 0; t < 4; ++t) {
      v16h bw = *(const v16h*)(W + (size_t)(n0 + t * 16 + ml) * ldw + k0 + h * 16);
#pragma unroll
      for (int u = 0; u < 2; ++u) acc[u][t] = wmma_f16(av[u], bw, acc[u][t]);
    }
  }

#pragma unroll
  for (int t = 0; t < 4; ++t) {
    const int n = n0 + t * 16 + ml;
    const float bv = bias[n];
#pragma unroll
    for (int u = 0; u < 2; ++u)
#pragma unroll
      for (int r = 0; r < 8; ++r) {
        const int m = m0 + u * 16 + h * 8 + r;
        const float v = acc[u][t][r] + bv;
        if (outh) outh[(size_t)m * ldo + n] = (_Float16)v;
        else      outf[(size_t)m * ldo + n] = v;
      }
  }
}

// --------------------------------------------- V transpose: vt[d][n] = v[n][d]
__global__ __launch_bounds__(256)
void k_transpose_v(const _Float16* __restrict__ qkvh, _Float16* __restrict__ vt) {
  __shared__ _Float16 tile[32][33];
  const int n0 = blockIdx.x * 32;
  const int d0 = blockIdx.y * 32;
  const int tx = threadIdx.x & 31;
  const int ty = threadIdx.x >> 5;
#pragma unroll
  for (int i = 0; i < 4; ++i) {
    const int nr = ty + i * 8;
    tile[nr][tx] = qkvh[(size_t)(n0 + nr) * TD + 2 * DIM + d0 + tx];
  }
  __syncthreads();
#pragma unroll
  for (int i = 0; i < 4; ++i) {
    const int dr = ty + i * 8;
    vt[(size_t)(d0 + dr) * N_TOK + n0 + tx] = tile[tx][dr];
  }
}

// ------------------------------------------------------- masked flash attention
// One wave per (64-query tile, head); 8 waves/block = 8 heads (head-group in
// blockIdx.y reads a disjoint column slice -> no duplicated K/V traffic).
// Computed transposed so softmax state is per-lane:
//   S^T = K_tile(16x64) * Q^T(64x16)      ctx^T = V^T(64x32) * P(32x16)
// 4 query sub-tiles share every K/V fragment (4x L2-traffic reduction).
__global__ __launch_bounds__(256, 1)
void k_attn(const _Float16* __restrict__ qkvh, const _Float16* __restrict__ vt,
            const int* __restrict__ cid, _Float16* __restrict__ ctxh) {
  const int q0   = blockIdx.x * 64;
  const int head = blockIdx.y * 8 + (threadIdx.x >> 5);
  const int lane = threadIdx.x & 31;
  const int n    = lane & 15;            // query column / frag row index
  const int h    = lane >> 4;
  const float scale = 0.125f;            // 1/sqrt(64)
  const float NEG = -1e30f;

  // Q^T B-fragments (loaded once): elems j = head-dim  c*32 + h*16 + j
  v16h bq[4][2];
  int cq[4];
#pragma unroll
  for (int s = 0; s < 4; ++s) {
    const _Float16* qrow = qkvh + (size_t)(q0 + s * 16 + n) * TD + head * HD;
    bq[s][0] = *(const v16h*)(qrow + h * 16);
    bq[s][1] = *(const v16h*)(qrow + 32 + h * 16);
    cq[s] = cid[q0 + s * 16 + n];
  }

  v8f acc[4][4] = {};
  const v8f vzero = {};
  float m_run[4] = {NEG, NEG, NEG, NEG};
  float l_run[4] = {0.f, 0.f, 0.f, 0.f};

  for (int kb = 0; kb < N_TOK; kb += 32) {
    if (kb + 32 < N_TOK) {  // prefetch next key tile (global_prefetch_b8)
      __builtin_prefetch(qkvh + (size_t)(kb + 32 + n) * TD + DIM + head * HD, 0, 1);
      __builtin_prefetch(qkvh + (size_t)(kb + 48 + n) * TD + DIM + head * HD, 0, 1);
    }

    // cluster ids of this lane's 8 key rows in each 16-key group
    const int4 ca = *(const int4*)(cid + kb + h * 8);
    const int4 cb = *(const int4*)(cid + kb + h * 8 + 4);
    const int4 cc = *(const int4*)(cid + kb + 16 + h * 8);
    const int4 cd = *(const int4*)(cid + kb + 16 + h * 8 + 4);
    const int ck0[8] = {ca.x, ca.y, ca.z, ca.w, cb.x, cb.y, cb.z, cb.w};
    const int ck1[8] = {cc.x, cc.y, cc.z, cc.w, cd.x, cd.y, cd.z, cd.w};

    // K tile A-fragments, shared by all 4 query sub-tiles
    v16h ak[2][2];
#pragma unroll
    for (int g = 0; g < 2; ++g) {
      const _Float16* krow =
          qkvh + (size_t)(kb + g * 16 + n) * TD + DIM + head * HD;
      ak[g][0] = *(const v16h*)(krow + h * 16);
      ak[g][1] = *(const v16h*)(krow + 32 + h * 16);
    }

    v16h pb[4];
#pragma unroll
    for (int s = 0; s < 4; ++s) {
      v8f t0 = wmma_f16(ak[0][1], bq[s][1], vzero);
      t0 = wmma_f16(ak[0][0], bq[s][0], t0);
      v8f t1 = wmma_f16(ak[1][1], bq[s][1], vzero);
      t1 = wmma_f16(ak[1][0], bq[s][0], t1);

      float s0[8], s1[8];
#pragma unroll
      for (int r = 0; r < 8; ++r) {
        s0[r] = (ck0[r] == cq[s]) ? t0[r] * scale : NEG;
        s1[r] = (ck1[r] == cq[s]) ? t1[r] * scale : NEG;
      }

      // online softmax per query (== per lane; both halves identical)
      float tmax = NEG;
#pragma unroll
      for (int r = 0; r < 8; ++r) tmax = fmaxf(tmax, fmaxf(s0[r], s1[r]));
      tmax = fmaxf(tmax, __shfl_xor(tmax, 16, 32));
      const float m_new = fmaxf(m_run[s], tmax);
      const float alpha = __expf(m_run[s] - m_new);
      float p0[8], p1[8], rsum = 0.f;
#pragma unroll
      for (int r = 0; r < 8; ++r) {
        p0[r] = __expf(s0[r] - m_new);
        p1[r] = __expf(s1[r] - m_new);
        rsum += p0[r] + p1[r];
      }
      rsum += __shfl_xor(rsum, 16, 32);
      l_run[s] = l_run[s] * alpha + rsum;
      m_run[s] = m_new;
#pragma unroll
      for (int c = 0; c < 4; ++c)
#pragma unroll
        for (int r = 0; r < 8; ++r) acc[s][c][r] *= alpha;

      // P B-fragment (32 keys x 16 queries): elems j = key h*16+j;
      // one cross-half shuffle per element routes the other half's rows.
#pragma unroll
      for (int r = 0; r < 8; ++r) {
        const float o0 = __shfl_xor(p0[r], 16, 32);
        const float o1 = __shfl_xor(p1[r], 16, 32);
        pb[s][r]     = (_Float16)((h == 0) ? p0[r] : o1);
        pb[s][r + 8] = (_Float16)((h == 0) ? o0 : p1[r]);
      }
    }

    // ctx^T += V^T-chunk(16 dims x 32 keys) * P   (V fragment reused 4x)
#pragma unroll
    for (int c = 0; c < 4; ++c) {
      const _Float16* vrow = vt + (size_t)(head * HD + c * 16 + n) * N_TOK + kb;
      v16h av = *(const v16h*)(vrow + h * 16);
#pragma unroll
      for (int s = 0; s < 4; ++s) acc[s][c] = wmma_f16(av, pb[s], acc[s][c]);
    }
  }

#pragma unroll
  for (int s = 0; s < 4; ++s) {
    const float inv = 1.f / l_run[s];
#pragma unroll
    for (int c = 0; c < 4; ++c)
#pragma unroll
      for (int r = 0; r < 8; ++r)
        ctxh[(size_t)(q0 + s * 16 + n) * DIM + head * HD + c * 16 + h * 8 + r] =
            (_Float16)(acc[s][c][r] * inv);
  }
}

// ---------------------------------------------- residual + LayerNorm epilogue
__global__ __launch_bounds__(256)
void k_ln(const float* __restrict__ x, const float* __restrict__ attn,
          const float* __restrict__ lnw, const float* __restrict__ lnb,
          float* __restrict__ out) {
  __shared__ float sh[256];
  const int row = blockIdx.x;
  const int tid = threadIdx.x;
  float v[4], s1 = 0.f, s2 = 0.f;
#pragma unroll
  for (int i = 0; i < 4; ++i) {
    const int j = tid + i * 256;
    v[i] = x[(size_t)row * DIM + j] + attn[(size_t)row * DIM + j];
    s1 += v[i];
    s2 += v[i] * v[i];
  }
  sh[tid] = s1; __syncthreads();
  for (int off = 128; off; off >>= 1) { if (tid < off) sh[tid] += sh[tid + off]; __syncthreads(); }
  const float mu = sh[0] * (1.f / DIM);
  __syncthreads();
  sh[tid] = s2; __syncthreads();
  for (int off = 128; off; off >>= 1) { if (tid < off) sh[tid] += sh[tid + off]; __syncthreads(); }
  const float var = sh[0] * (1.f / DIM) - mu * mu;
  const float rs = rsqrtf(var + 1e-5f);
#pragma unroll
  for (int i = 0; i < 4; ++i) {
    const int j = tid + i * 256;
    out[(size_t)row * DIM + j] = (v[i] - mu) * rs * lnw[j] + lnb[j];
  }
}

// ---------------------------------------------------------------------- launch
extern "C" void kernel_launch(void* const* d_in, const int* in_sizes, int n_in,
                              void* d_out, int out_size, void* d_ws, size_t ws_size,
                              hipStream_t stream) {
  (void)in_sizes; (void)n_in; (void)out_size; (void)ws_size;
  const float* x    = (const float*)d_in[0];
  const int*   cidp = (const int*)d_in[1];   // jax int64 demoted to int32 (no x64)
  const float* wqkv = (const float*)d_in[2];
  const float* bqkv = (const float*)d_in[3];
  const float* wout = (const float*)d_in[4];
  const float* bout = (const float*)d_in[5];
  const float* lnw  = (const float*)d_in[6];
  const float* lnb  = (const float*)d_in[7];
  float* out = (float*)d_out;

  char* ws = (char*)d_ws;
  _Float16* x_h    = (_Float16*)ws; ws += (size_t)N_TOK * DIM * 2;  //  8 MB
  _Float16* wqkv_h = (_Float16*)ws; ws += (size_t)TD * DIM * 2;     //  6 MB
  _Float16* wout_h = (_Float16*)ws; ws += (size_t)DIM * DIM * 2;    //  2 MB
  _Float16* qkv_h  = (_Float16*)ws; ws += (size_t)N_TOK * TD * 2;   // 24 MB
  _Float16* vt_h   = (_Float16*)ws; ws += (size_t)DIM * N_TOK * 2;  //  8 MB
  _Float16* ctx_h  = (_Float16*)ws; ws += (size_t)N_TOK * DIM * 2;  //  8 MB
  float*    attn   = (float*)ws;                                    // 16 MB

  k_cvt<<<1024, 256, 0, stream>>>(x, x_h, N_TOK * DIM);
  k_cvt<<<1024, 256, 0, stream>>>(wqkv, wqkv_h, TD * DIM);
  k_cvt<<<512, 256, 0, stream>>>(wout, wout_h, DIM * DIM);

  dim3 g1(N_TOK / 32, TD / 512);
  k_gemm<<<g1, 256, 0, stream>>>(x_h, DIM, wqkv_h, DIM, bqkv, qkv_h, nullptr, TD, DIM);

  dim3 gt(N_TOK / 32, DIM / 32);
  k_transpose_v<<<gt, 256, 0, stream>>>(qkv_h, vt_h);

  dim3 ga(N_TOK / 64, NH / 8);
  k_attn<<<ga, 256, 0, stream>>>(qkv_h, vt_h, cidp, ctx_h);

  dim3 g2(N_TOK / 32, DIM / 512);
  k_gemm<<<g2, 256, 0, stream>>>(ctx_h, DIM, wout_h, DIM, bout, nullptr, attn, DIM, DIM);

  k_ln<<<N_TOK, 256, 0, stream>>>(x, attn, lnw, lnb, out);
}
